// Unet5ContextualBneck_4750233829614
// MI455X (gfx1250) — matrix operand order repaired and checked by reading, not compiled, not run on hardware
//
#include <hip/hip_runtime.h>
#include <hip/hip_bf16.h>

// ---------------------------------------------------------------------------
// CDNA5 (gfx1250) implementation of the language-conditioned U-Net.
// Every 5x5 conv / transposed-conv is an implicit GEMM on
// v_wmma_f32_16x16x32_f16 (wave32), staged through LDS.
// ---------------------------------------------------------------------------

typedef __attribute__((ext_vector_type(16))) _Float16 v16h;
typedef __attribute__((ext_vector_type(8)))  float    v8f;
typedef unsigned int u32x4 __attribute__((ext_vector_type(4)));

union V16H { v16h v; _Float16 h[16]; u32x4 q[2]; };

static constexpr int KP     = 224;     // padded K per 8-channel group (8*25 -> 224)
static constexpr int CHUNKS = KP / 32; // WMMA K=32 steps per group

// ---------------------------------------------------------------------------
// conv5x5 (stride2,pad2) / deconv5x5 (stride2,pad2,outpad1) as implicit GEMM.
// mode 0: y[oy,ox] += x[2oy-2+ky, 2ox-2+kx] * w[n][c][ky][kx]      (OIHW)
// mode 1: y[oy,ox] += x[(oy+2-ky)/2, (ox+2-kx)/2] * w[c][n][ky][kx] (I,O,kh,kw)
//         tap valid iff (oy+2-ky) even & in range (gather form, no flip).
// Channel-concat inputs: c < cinA -> srcA, else srcB.
// Block = 4 wave32s: each wave owns one 16-pixel M tile; B tile is shared.
// ---------------------------------------------------------------------------
__global__ __launch_bounds__(128) void conv5x5_wmma_kernel(
    const float* __restrict__ srcA, const float* __restrict__ srcB,
    int cinA, int cinB,
    const float* __restrict__ wgt, const float* __restrict__ bias,
    float* __restrict__ dst,
    int Hin, int Win, int Hout, int Wout, int Cout,
    int mode, int act)
{
    __shared__ __attribute__((aligned(32))) _Float16 Alds[4][16 * KP];
    __shared__ __attribute__((aligned(32))) _Float16 Blds[16 * KP];

    const int lane   = threadIdx.x & 31;
    const int wave   = threadIdx.x >> 5;
    const int b      = blockIdx.z;
    const int ntile  = blockIdx.y;
    const int HW     = Hout * Wout;
    const int mtiles = HW >> 4;
    const int mtile  = blockIdx.x * 4 + wave;
    const int cin    = cinA + cinB;

    v8f acc = {};

    for (int cg0 = 0; cg0 < cin; cg0 += 8) {
        const int ccnt = (cin - cg0 < 8) ? (cin - cg0) : 8;
        const int Kc   = ccnt * 25;

        // ---- stage A (im2col tile), one tile per wave --------------------
        if (mtile < mtiles) {
            for (int idx = lane; idx < 16 * KP; idx += 32) {
                const int m = idx / KP;
                const int k = idx - m * KP;
                _Float16 val = (_Float16)0.0f;
                if (k < Kc) {
                    const int cl = k / 25, t = k - cl * 25;
                    const int ky = t / 5,  kx = t - ky * 5;
                    const int c  = cg0 + cl;
                    const int p  = mtile * 16 + m;
                    const int oy = p / Wout, ox = p - oy * Wout;
                    int iy = 0, ix = 0;
                    bool ok;
                    if (mode == 0) {
                        iy = oy * 2 - 2 + ky;
                        ix = ox * 2 - 2 + kx;
                        ok = (iy >= 0) & (iy < Hin) & (ix >= 0) & (ix < Win);
                    } else {
                        const int ty = oy + 2 - ky, tx = ox + 2 - kx;
                        ok = (ty >= 0) & (tx >= 0) & !(ty & 1) & !(tx & 1);
                        iy = ty >> 1; ix = tx >> 1;
                        ok = ok & (iy < Hin) & (ix < Win);
                    }
                    if (ok) {
                        const float* s; int cn, cb;
                        if (c < cinA) { s = srcA; cn = c;        cb = cinA; }
                        else          { s = srcB; cn = c - cinA; cb = cinB; }
                        val = (_Float16)s[(((size_t)b * cb + cn) * Hin + iy) * Win + ix];
                    }
                }
                Alds[wave][m * KP + k] = val;
            }
        }
        // ---- stage B (weight tile), shared by all 4 waves ----------------
        for (int idx = threadIdx.x; idx < 16 * KP; idx += 128) {
            const int n  = idx / KP;
            const int k  = idx - n * KP;
            const int ng = ntile * 16 + n;
            _Float16 val = (_Float16)0.0f;
            if ((k < Kc) && (ng < Cout)) {
                const int cl = k / 25, t = k - cl * 25;
                const int ky = t / 5,  kx = t - ky * 5;
                const int c  = cg0 + cl;
                size_t widx;
                if (mode == 0) widx = (((size_t)ng * cin  + c ) * 5 + ky) * 5 + kx;
                else           widx = (((size_t)c  * Cout + ng) * 5 + ky) * 5 + kx;
                val = (_Float16)wgt[widx];
            }
            Blds[n * KP + k] = val;
        }
        __syncthreads();

        // ---- WMMA over this K group --------------------------------------
        if (mtile < mtiles) {
            const int half = lane >> 4;
            const int row  = lane & 15;   // M row for A, N col for B
            for (int kc = 0; kc < CHUNKS; ++kc) {
                V16H a, bm;
                // A 16x32 f16: lane half selects K sub-blocks {half*8..} and {16+half*8..}
                const _Float16* pa = &Alds[wave][row * KP + kc * 32];
                a.q[0] = *reinterpret_cast<const u32x4*>(pa + half * 8);
                a.q[1] = *reinterpret_cast<const u32x4*>(pa + 16 + half * 8);
                // B 32x16 f16: lane half selects K sub-blocks {half*8..} and {16+half*8..}
                const _Float16* pb = &Blds[row * KP + kc * 32 + half * 8];
                bm.q[0] = *reinterpret_cast<const u32x4*>(pb);
                bm.q[1] = *reinterpret_cast<const u32x4*>(pb + 16);
                acc = __builtin_amdgcn_wmma_f32_16x16x32_f16(
                        false, a.v, false, bm.v, (short)0, acc, false, false);
            }
        }
        __syncthreads();
    }

    // ---- epilogue: C/D layout (VGPR r -> M = r + 8*(lane>=16), lane%16 -> N)
    if (mtile < mtiles) {
        const int ncol  = lane & 15;
        const int mbase = (lane >> 4) * 8;
        const int ng    = ntile * 16 + ncol;
        if (ng < Cout) {
            const float bv = bias[ng];
            for (int r = 0; r < 8; ++r) {
                const int p  = mtile * 16 + mbase + r;
                const int oy = p / Wout, ox = p - oy * Wout;
                float v = acc[r] + bv;
                if (act) v = v > 0.0f ? v : 0.01f * v;
                dst[(((size_t)b * Cout + ng) * Hout + oy) * Wout + ox] = v;
            }
        }
    }
}

// ---------------------------------------------------------------------------
// InstanceNorm2d(affine=False), eps=1e-5, biased variance. One block per
// (b,c) plane, in place.
// ---------------------------------------------------------------------------
__global__ __launch_bounds__(256) void inorm_kernel(float* __restrict__ x, int HW)
{
    const size_t base = (size_t)blockIdx.x * HW;
    float s = 0.0f, s2 = 0.0f;
    for (int i = threadIdx.x; i < HW; i += 256) {
        const float v = x[base + i];
        s += v; s2 += v * v;
    }
    __shared__ float r1[256], r2[256];
    r1[threadIdx.x] = s; r2[threadIdx.x] = s2;
    __syncthreads();
    for (int off = 128; off > 0; off >>= 1) {
        if (threadIdx.x < off) {
            r1[threadIdx.x] += r1[threadIdx.x + off];
            r2[threadIdx.x] += r2[threadIdx.x + off];
        }
        __syncthreads();
    }
    const float mean = r1[0] / (float)HW;
    const float var  = r2[0] / (float)HW - mean * mean;
    const float rstd = rsqrtf(var + 1e-5f);
    for (int i = threadIdx.x; i < HW; i += 256)
        x[base + i] = (x[base + i] - mean) * rstd;
}

// ---------------------------------------------------------------------------
// FiLM: out[b,o,p] = sum_c x[b,c,p] * wb[b,o,c]  (per-sample 1x1 conv)
// ---------------------------------------------------------------------------
__global__ __launch_bounds__(256) void film_kernel(
    const float* __restrict__ x, const float* __restrict__ wb,
    float* __restrict__ out, int Cin, int Cout, int HW)
{
    const int b = blockIdx.z;
    const int o = blockIdx.y;
    const int p = blockIdx.x * 256 + threadIdx.x;
    if (p >= HW) return;
    const float* w  = wb + ((size_t)b * Cout + o) * Cin;
    const float* xp = x  + (size_t)b * Cin * HW + p;
    float acc = 0.0f;
    for (int c = 0; c < Cin; ++c)
        acc = fmaf(xp[(size_t)c * HW], w[c], acc);
    out[((size_t)b * Cout + o) * HW + p] = acc;
}

// ---------------------------------------------------------------------------
// Language weights: v = emb @ W^T + b; L2-normalize whole vector (eps 1e-12).
// One block per sample; N = co*ci <= 1024, EMB = 256.
// ---------------------------------------------------------------------------
__global__ __launch_bounds__(256) void lang_kernel(
    const float* __restrict__ emb, const float* __restrict__ W,
    const float* __restrict__ bias, float* __restrict__ out, int N)
{
    const int b = blockIdx.x;
    __shared__ float ebuf[256];
    __shared__ float vbuf[1024];
    __shared__ float red[256];
    ebuf[threadIdx.x] = emb[b * 256 + threadIdx.x];
    __syncthreads();
    float ss = 0.0f;
    for (int j = threadIdx.x; j < N; j += 256) {
        const float* wr = W + (size_t)j * 256;
        float acc = bias[j];
        for (int e = 0; e < 256; ++e) acc = fmaf(ebuf[e], wr[e], acc);
        vbuf[j] = acc;
        ss += acc * acc;
    }
    red[threadIdx.x] = ss;
    __syncthreads();
    for (int off = 128; off > 0; off >>= 1) {
        if (threadIdx.x < off) red[threadIdx.x] += red[threadIdx.x + off];
        __syncthreads();
    }
    const float rn = 1.0f / fmaxf(sqrtf(red[0]), 1e-12f);
    for (int j = threadIdx.x; j < N; j += 256)
        out[(size_t)b * N + j] = vbuf[j] * rn;
}

// ---------------------------------------------------------------------------
extern "C" void kernel_launch(void* const* d_in, const int* in_sizes, int n_in,
                              void* d_out, int out_size, void* d_ws, size_t ws_size,
                              hipStream_t stream)
{
    (void)in_sizes; (void)n_in; (void)out_size; (void)ws_size;
    const float* x   = (const float*)d_in[0];
    const float* emb = (const float*)d_in[1];
    const float* c1w = (const float*)d_in[2];  const float* c1b = (const float*)d_in[3];
    const float* c2w = (const float*)d_in[4];  const float* c2b = (const float*)d_in[5];
    const float* c3w = (const float*)d_in[6];  const float* c3b = (const float*)d_in[7];
    const float* c4w = (const float*)d_in[8];  const float* c4b = (const float*)d_in[9];
    const float* c5w = (const float*)d_in[10]; const float* c5b = (const float*)d_in[11];
    const float* d1w = (const float*)d_in[12]; const float* d1b = (const float*)d_in[13];
    const float* d2w = (const float*)d_in[14]; const float* d2b = (const float*)d_in[15];
    const float* d3w = (const float*)d_in[16]; const float* d3b = (const float*)d_in[17];
    const float* d4w = (const float*)d_in[18]; const float* d4b = (const float*)d_in[19];
    const float* d5w = (const float*)d_in[20]; const float* d5b = (const float*)d_in[21];
    const float* l1w = (const float*)d_in[22]; const float* l1b = (const float*)d_in[23];
    const float* l2w = (const float*)d_in[24]; const float* l2b = (const float*)d_in[25];
    const float* l3w = (const float*)d_in[26]; const float* l3b = (const float*)d_in[27];
    const float* l4w = (const float*)d_in[28]; const float* l4b = (const float*)d_in[29];
    const float* l5w = (const float*)d_in[30]; const float* l5b = (const float*)d_in[31];

    const int B = 8;
    float* ws = (float*)d_ws;
    size_t off = 0;
    auto alloc = [&](size_t n) { float* p = ws + off; off += n; return p; };
    float* wb1 = alloc(B * 512);
    float* wb2 = alloc(B * 512);
    float* wb3 = alloc(B * 512);
    float* wb4 = alloc(B * 512);
    float* wb5 = alloc(B * 1024);
    float* x1  = alloc((size_t)B * 32 * 128 * 128);
    float* x2  = alloc((size_t)B * 32 * 64 * 64);
    float* x3  = alloc((size_t)B * 32 * 32 * 32);
    float* x4  = alloc((size_t)B * 32 * 16 * 16);
    float* x5  = alloc((size_t)B * 32 * 8 * 8);
    float* x1f = alloc((size_t)B * 16 * 128 * 128);
    float* x2f = alloc((size_t)B * 16 * 64 * 64);
    float* x3f = alloc((size_t)B * 16 * 32 * 32);
    float* x4f = alloc((size_t)B * 16 * 16 * 16);
    float* x5f = alloc((size_t)B * 32 * 8 * 8);
    float* x6  = alloc((size_t)B * 32 * 16 * 16);
    float* x7  = alloc((size_t)B * 32 * 32 * 32);
    float* x8  = alloc((size_t)B * 32 * 64 * 64);
    float* x9  = alloc((size_t)B * 256 * 128 * 128);

    // -------- language-conditioned 1x1 weights --------------------------
    lang_kernel<<<B, 256, 0, stream>>>(emb, l1w, l1b, wb1, 512);
    lang_kernel<<<B, 256, 0, stream>>>(emb, l2w, l2b, wb2, 512);
    lang_kernel<<<B, 256, 0, stream>>>(emb, l3w, l3b, wb3, 512);
    lang_kernel<<<B, 256, 0, stream>>>(emb, l4w, l4b, wb4, 512);
    lang_kernel<<<B, 256, 0, stream>>>(emb, l5w, l5b, wb5, 1024);

    auto conv = [&](const float* sA, const float* sB, int cA, int cB,
                    const float* w, const float* bs, float* dst,
                    int Hin, int Hout, int Cout, int mode, int act) {
        const int mtiles = (Hout * Hout) >> 4;
        dim3 grid((mtiles + 3) / 4, Cout >= 16 ? Cout / 16 : 1, B);
        conv5x5_wmma_kernel<<<grid, 128, 0, stream>>>(
            sA, sB, cA, cB, w, bs, dst, Hin, Hin, Hout, Hout, Cout, mode, act);
    };
    auto inorm = [&](float* p, int C, int HW) {
        inorm_kernel<<<B * C, 256, 0, stream>>>(p, HW);
    };
    auto film = [&](const float* src, const float* wb, float* dst,
                    int Cin, int Cout, int HW) {
        dim3 grid((HW + 255) / 256, Cout, B);
        film_kernel<<<grid, 256, 0, stream>>>(src, wb, dst, Cin, Cout, HW);
    };

    // -------- encoder ----------------------------------------------------
    conv(x,  nullptr, 3,  0, c1w, c1b, x1, 256, 128, 32, 0, 1); inorm(x1, 32, 128 * 128);
    conv(x1, nullptr, 32, 0, c2w, c2b, x2, 128,  64, 32, 0, 1); inorm(x2, 32,  64 *  64);
    conv(x2, nullptr, 32, 0, c3w, c3b, x3,  64,  32, 32, 0, 1); inorm(x3, 32,  32 *  32);
    conv(x3, nullptr, 32, 0, c4w, c4b, x4,  32,  16, 32, 0, 1); inorm(x4, 32,  16 *  16);
    conv(x4, nullptr, 32, 0, c5w, c5b, x5,  16,   8, 32, 0, 1);   // lrelu only

    // -------- FiLM conditioning ------------------------------------------
    film(x1, wb1, x1f, 32, 16, 128 * 128);
    film(x2, wb2, x2f, 32, 16,  64 *  64);
    film(x3, wb3, x3f, 32, 16,  32 *  32);
    film(x4, wb4, x4f, 32, 16,  16 *  16);
    film(x5, wb5, x5f, 32, 32,   8 *   8);

    // -------- decoder with skip concats ----------------------------------
    conv(x5f, nullptr, 32,  0,  d1w, d1b, x6,    8,  16,  32, 1, 1);              // x6
    conv(x4f, x6,      16, 32,  d2w, d2b, x7,   16,  32,  32, 1, 1); inorm(x7, 32,   32 * 32);
    conv(x3f, x7,      16, 32,  d3w, d3b, x8,   32,  64,  32, 1, 1); inorm(x8, 32,   64 * 64);
    conv(x2f, x8,      16, 32,  d4w, d4b, x9,   64, 128, 256, 1, 1); inorm(x9, 256, 128 * 128);
    conv(x1f, x9,      16, 256, d5w, d5b, (float*)d_out, 128, 256, 2, 1, 0);      // final, linear
}
